// EngramUpsampler_3753801417053
// MI455X (gfx1250) — matrix operand order repaired and checked by reading, not compiled, not run on hardware
//
#include <hip/hip_runtime.h>
#include <math.h>

// ---------------------------------------------------------------------------
// EngramUpsampler fused kernel for gfx1250 (MI455X)
//   x = repeat_interleave(engrams, 32) + pos
//   h = gelu(x @ w1^T + b1)        (WMMA f16 -> f32 acc)
//   y = x + h @ w2^T + b2          (WMMA f16 -> f32 acc)
//   out = LayerNorm(y) * gamma + beta
// M_TILE = 32 rows/workgroup: each B-fragment feeds 2 WMMAs (2 M-tiles),
// halving L2->WGP weight traffic per FLOP vs the 16-row version.
// ---------------------------------------------------------------------------

typedef __attribute__((ext_vector_type(16))) _Float16 v16h;
typedef __attribute__((ext_vector_type(8)))  _Float16 v8h;
typedef __attribute__((ext_vector_type(8)))  float    v8f;

#define D_DIM   1024
#define M_TILE  32
#define PAD     8                  // _Float16 elements of row padding
#define ROW     (D_DIM + PAD)      // 1032
#define NKT     (D_DIM / 32)       // 32 k-tiles per GEMM
#define NNT     (D_DIM / 16)       // 64 n-tiles per GEMM

// ---------------------------------------------------------------------------
// Pack weight matrix (fp32 row-major w[e][d]) into WMMA B-fragment order, f16.
// Tile (nt, kt) -> 512 f16 at offset (nt*NKT + kt)*512.
// Within a tile: element (lane, j):
//   n = lane & 15
//   k = kt*32 + (j & 7) + (j >= 8 ? 16 : 0) + (lane >= 16 ? 8 : 0)   (table order, matches A)
//   value = w[nt*16 + n][k]
// Main-loop load per lane: 16 contiguous f16 (32B) -> perfectly coalesced.
// ---------------------------------------------------------------------------
__global__ __launch_bounds__(256) void pack_w_f16(const float* __restrict__ w,
                                                  _Float16* __restrict__ wp) {
    int idx    = blockIdx.x * 256 + threadIdx.x;   // 0 .. 2^20-1
    int tile   = idx >> 9;
    int within = idx & 511;
    int lane   = within >> 4;
    int j      = within & 15;
    int nt     = tile / NKT;
    int kt     = tile % NKT;
    int e      = nt * 16 + (lane & 15);
    int k      = kt * 32 + (j & 7) + ((j & 8) ? 16 : 0) + ((lane & 16) ? 8 : 0);
    wp[idx] = (_Float16)w[e * D_DIM + k];
}

// ---------------------------------------------------------------------------
// Fused per-32-row-tile kernel. 256 threads = 8 wave32.
// Dynamic LDS (264192 B -> 1 workgroup / 320KB WGP):
//   xF : float    [32][ROW]   (x, overwritten in place by y)     132096 B
//   xA : _Float16 [32][ROW]   (f16 copy of x, A-matrix GEMM1)     66048 B
//   hA : _Float16 [32][ROW]   (f16 gelu output, A-matrix GEMM2)   66048 B
// ---------------------------------------------------------------------------
extern __shared__ char smem_raw[];

__global__ __launch_bounds__(256) void engram_fused_mlp(
    const float* __restrict__ engrams,   // (4,16,16,1024)
    const float* __restrict__ pos,       // (512,1024)
    const _Float16* __restrict__ w1p,    // packed (1<<20)
    const _Float16* __restrict__ w2p,    // packed (1<<20)
    const float* __restrict__ b1,
    const float* __restrict__ b2,
    const float* __restrict__ gamma,
    const float* __restrict__ beta,
    float* __restrict__ out)             // (32768,1024)
{
    float*    xF = (float*)smem_raw;
    _Float16* xA = (_Float16*)(smem_raw + M_TILE * ROW * sizeof(float));
    _Float16* hA = xA + M_TILE * ROW;

    const int tid = threadIdx.x;
    const int r0  = blockIdx.x * M_TILE;   // global row base
    const int bn  = r0 >> 9;               // which (b, nw) slice
    const int w0  = r0 & 511;              // window position base

    // ---------------- Phase 0: x = engram_repeat + pos (fp32 + f16) ----------
    for (int i = tid; i < M_TILE * D_DIM; i += 256) {
        int m = i >> 10;
        int e = i & (D_DIM - 1);
        int w = w0 + m;
        float v = engrams[(((bn << 4) + (w >> 5)) << 10) + e] + pos[(w << 10) + e];
        xF[m * ROW + e] = v;
        xA[m * ROW + e] = (_Float16)v;
    }
    __syncthreads();

    const int lane  = tid & 31;
    const int wv    = tid >> 5;
    const int mrow  = lane & 15;
    const int khalf = (lane & 16) ? 8 : 0;   // A-frag interleaved K layout (ISA 7.12.2)
    const int hi    = (lane & 16) ? 8 : 0;   // C/D layout: lanes 16-31 hold M=8..15
    const int ncol  = lane & 15;

    // ---------------- Phase 1: h = gelu(x @ w1^T + b1) -----------------------
    // Wave owns 8 N-tiles (nt = wv + 8*t). Per pass: 2 N-tiles x 2 M-tiles ->
    // 4 independent WMMA chains; each A-frag and each B-frag feeds 2 WMMAs.
    #pragma unroll 1
    for (int s = 0; s < 4; ++s) {
        const int nt0 = wv + 8 * (2 * s);
        const int nt1 = wv + 8 * (2 * s + 1);
        v8f acc[2][2] = {};                   // [mt][ntq]
        const _Float16* __restrict__ bp0 = w1p + (size_t)nt0 * NKT * 512 + lane * 16;
        const _Float16* __restrict__ bp1 = w1p + (size_t)nt1 * NKT * 512 + lane * 16;
        const _Float16* __restrict__ ap0 = xA + (mrow     ) * ROW + khalf;
        const _Float16* __restrict__ ap1 = xA + (mrow + 16) * ROW + khalf;
        for (int kt = 0; kt < NKT; ++kt) {
            if (kt + 4 < NKT) {
                __builtin_prefetch(bp0 + (kt + 4) * 512, 0, 3);
                __builtin_prefetch(bp1 + (kt + 4) * 512, 0, 3);
            }
            v8h a0lo = *(const v8h*)(ap0 + kt * 32);
            v8h a0hi = *(const v8h*)(ap0 + kt * 32 + 16);
            v8h a1lo = *(const v8h*)(ap1 + kt * 32);
            v8h a1hi = *(const v8h*)(ap1 + kt * 32 + 16);
            v16h a0 = __builtin_shufflevector(a0lo, a0hi, 0,1,2,3,4,5,6,7,8,9,10,11,12,13,14,15);
            v16h a1 = __builtin_shufflevector(a1lo, a1hi, 0,1,2,3,4,5,6,7,8,9,10,11,12,13,14,15);
            v16h b0 = *(const v16h*)(bp0 + kt * 512);
            v16h b1 = *(const v16h*)(bp1 + kt * 512);
            acc[0][0] = __builtin_amdgcn_wmma_f32_16x16x32_f16(false, a0, false, b0,
                                                               (short)0, acc[0][0], false, false);
            acc[1][0] = __builtin_amdgcn_wmma_f32_16x16x32_f16(false, a1, false, b0,
                                                               (short)0, acc[1][0], false, false);
            acc[0][1] = __builtin_amdgcn_wmma_f32_16x16x32_f16(false, a0, false, b1,
                                                               (short)0, acc[0][1], false, false);
            acc[1][1] = __builtin_amdgcn_wmma_f32_16x16x32_f16(false, a1, false, b1,
                                                               (short)0, acc[1][1], false, false);
        }
        #pragma unroll
        for (int mt = 0; mt < 2; ++mt) {
            #pragma unroll
            for (int q = 0; q < 2; ++q) {
                const int col  = ((q == 0) ? nt0 : nt1) * 16 + ncol;
                const float bias = b1[col];
                #pragma unroll
                for (int g = 0; g < 8; ++g) {
                    int m = mt * 16 + g + hi;
                    float v = acc[mt][q][g] + bias;
                    v = 0.5f * v * (1.0f + erff(v * 0.70710678118654752f));
                    hA[m * ROW + col] = (_Float16)v;
                }
            }
        }
    }
    __syncthreads();

    // ---------------- Phase 2: y = x + h @ w2^T + b2 (in place over xF) ------
    #pragma unroll 1
    for (int s = 0; s < 4; ++s) {
        const int nt0 = wv + 8 * (2 * s);
        const int nt1 = wv + 8 * (2 * s + 1);
        v8f acc[2][2] = {};
        const _Float16* __restrict__ bp0 = w2p + (size_t)nt0 * NKT * 512 + lane * 16;
        const _Float16* __restrict__ bp1 = w2p + (size_t)nt1 * NKT * 512 + lane * 16;
        const _Float16* __restrict__ ap0 = hA + (mrow     ) * ROW + khalf;
        const _Float16* __restrict__ ap1 = hA + (mrow + 16) * ROW + khalf;
        for (int kt = 0; kt < NKT; ++kt) {
            if (kt + 4 < NKT) {
                __builtin_prefetch(bp0 + (kt + 4) * 512, 0, 3);
                __builtin_prefetch(bp1 + (kt + 4) * 512, 0, 3);
            }
            v8h a0lo = *(const v8h*)(ap0 + kt * 32);
            v8h a0hi = *(const v8h*)(ap0 + kt * 32 + 16);
            v8h a1lo = *(const v8h*)(ap1 + kt * 32);
            v8h a1hi = *(const v8h*)(ap1 + kt * 32 + 16);
            v16h a0 = __builtin_shufflevector(a0lo, a0hi, 0,1,2,3,4,5,6,7,8,9,10,11,12,13,14,15);
            v16h a1 = __builtin_shufflevector(a1lo, a1hi, 0,1,2,3,4,5,6,7,8,9,10,11,12,13,14,15);
            v16h b0 = *(const v16h*)(bp0 + kt * 512);
            v16h b1 = *(const v16h*)(bp1 + kt * 512);
            acc[0][0] = __builtin_amdgcn_wmma_f32_16x16x32_f16(false, a0, false, b0,
                                                               (short)0, acc[0][0], false, false);
            acc[1][0] = __builtin_amdgcn_wmma_f32_16x16x32_f16(false, a1, false, b0,
                                                               (short)0, acc[1][0], false, false);
            acc[0][1] = __builtin_amdgcn_wmma_f32_16x16x32_f16(false, a0, false, b1,
                                                               (short)0, acc[0][1], false, false);
            acc[1][1] = __builtin_amdgcn_wmma_f32_16x16x32_f16(false, a1, false, b1,
                                                               (short)0, acc[1][1], false, false);
        }
        #pragma unroll
        for (int mt = 0; mt < 2; ++mt) {
            #pragma unroll
            for (int q = 0; q < 2; ++q) {
                const int col  = ((q == 0) ? nt0 : nt1) * 16 + ncol;
                const float bias = b2[col];
                #pragma unroll
                for (int g = 0; g < 8; ++g) {
                    int m = mt * 16 + g + hi;
                    xF[m * ROW + col] += acc[mt][q][g] + bias;   // residual: x already there
                }
            }
        }
    }
    __syncthreads();

    // ---------------- Phase 3: LayerNorm over D, coalesced fp32 store --------
    #pragma unroll 1
    for (int rr = 0; rr < 4; ++rr) {
        const int row = wv * 4 + rr;
        float s = 0.0f, s2 = 0.0f;
        #pragma unroll
        for (int i = 0; i < 32; ++i) {
            float v = xF[row * ROW + i * 32 + lane];
            s += v; s2 += v * v;
        }
        #pragma unroll
        for (int off = 16; off > 0; off >>= 1) {
            s  += __shfl_xor(s,  off, 32);
            s2 += __shfl_xor(s2, off, 32);
        }
        const float mu  = s * (1.0f / 1024.0f);
        const float var = s2 * (1.0f / 1024.0f) - mu * mu;
        const float rs  = rsqrtf(var + 1e-5f);
        float* __restrict__ orow = out + (size_t)(r0 + row) * D_DIM;
        #pragma unroll
        for (int i = 0; i < 32; ++i) {
            int e = i * 32 + lane;
            orow[e] = (xF[row * ROW + e] - mu) * rs * gamma[e] + beta[e];
        }
    }
}

// ---------------------------------------------------------------------------
extern "C" void kernel_launch(void* const* d_in, const int* in_sizes, int n_in,
                              void* d_out, int out_size, void* d_ws, size_t ws_size,
                              hipStream_t stream) {
    const float* engrams = (const float*)d_in[0];
    const float* pos     = (const float*)d_in[1];
    const float* w1      = (const float*)d_in[2];
    const float* b1      = (const float*)d_in[3];
    const float* w2      = (const float*)d_in[4];
    const float* b2      = (const float*)d_in[5];
    const float* gamma   = (const float*)d_in[6];
    const float* beta    = (const float*)d_in[7];
    float* out = (float*)d_out;

    _Float16* w1p = (_Float16*)d_ws;              // 2 MB
    _Float16* w2p = w1p + (1 << 20);              // 2 MB

    // Pack both weight matrices into WMMA B-fragment order (f16).
    pack_w_f16<<<dim3(4096), dim3(256), 0, stream>>>(w1, w1p);
    pack_w_f16<<<dim3(4096), dim3(256), 0, stream>>>(w2, w2p);

    const size_t lds_bytes = (size_t)M_TILE * ROW * sizeof(float)       // xF / y
                           + (size_t)M_TILE * ROW * sizeof(_Float16)    // xA
                           + (size_t)M_TILE * ROW * sizeof(_Float16);   // hA
    (void)hipFuncSetAttribute((const void*)engram_fused_mlp,
                              hipFuncAttributeMaxDynamicSharedMemorySize,
                              (int)lds_bytes);

    // 32768 rows / 32 rows per block = 1024 blocks, 256 threads (8 wave32).
    engram_fused_mlp<<<dim3(1024), dim3(256), lds_bytes, stream>>>(
        engrams, pos, w1p, w2p, b1, b2, gamma, beta, out);
}
